// LSTM_30356828848341
// MI455X (gfx1250) — matrix operand order repaired and checked by reading, compiled, not run on hardware
//
#include <hip/hip_runtime.h>
#include <hip/hip_bf16.h>
#include <math.h>

// Problem dims (match reference)
#define B_  32
#define T_  2048
#define D_  512
#define H_  512
#define G_  2048          // 4*H
#define BT_ (B_*T_)
#define L_  2
#define NWG_SCAN 16       // workgroups in persistent scan (each owns 32 hidden units)

typedef __bf16 bf16_t;
typedef __attribute__((ext_vector_type(16))) __bf16 v16bf;
typedef __attribute__((ext_vector_type(8)))  float  v8f;
typedef __attribute__((ext_vector_type(4)))  unsigned int v4u;
typedef __attribute__((ext_vector_type(8)))  int v8i;
typedef __attribute__((ext_vector_type(4)))  int v4i;

union AFrag { v16bf v; bf16_t h[16]; };
union CFrag { v8f  v; float  f[8];  };

__device__ __forceinline__ v8f vzero8() {
  v8f z = {0.f,0.f,0.f,0.f,0.f,0.f,0.f,0.f};
  return z;
}

// CDNA5 wave32 WMMA 16-bit A-matrix (16x32) element->K map.
__device__ __forceinline__ int amapK(int e, int hf) {
  return (e & 7) + ((e >> 3) << 4) + (hf << 3);
}
// 16-bit B-matrix (32x16): lane n holds column n&15; K = (e&7) + 16*(e>>3) + 8*hf
__device__ __forceinline__ int bmapK(int e, int hf) {
  return (e & 7) + ((e >> 3) << 4) + (hf << 3);
}

__device__ __forceinline__ float sigm(float x) {
  return 1.0f / (1.0f + __expf(-x));
}

// Async global->LDS copy, 16B per lane. INST_OFFSET is added to BOTH the LDS
// destination and the global source address (ISA 08_async_tensor §4.4).
#define ASYNC_CP16(ldsoff, gaddr, imm)                                   \
  asm volatile("global_load_async_to_lds_b128 %0, %1, off offset:" #imm \
               :: "v"(ldsoff), "v"(gaddr) : "memory")
#define WAIT_ASYNC(n) asm volatile("s_wait_asynccnt " #n ::: "memory")

__device__ __forceinline__ unsigned lds_off(const void* p) {
  // generic LDS pointer: low 32 bits are the LDS byte address (ISA §10.2)
  return (unsigned)(size_t)p;
}

// ---------------------------------------------------------------- utilities
__global__ void k_cvt_bf16(const float* __restrict__ src,
                           bf16_t* __restrict__ dst, int n) {
  int i = blockIdx.x * blockDim.x + threadIdx.x;
  if (i < n) dst[i] = (bf16_t)src[i];
}

__global__ void k_bias_sum(const float* __restrict__ a,
                           const float* __restrict__ b,
                           float* __restrict__ o, int n) {
  int i = blockIdx.x * blockDim.x + threadIdx.x;
  if (i < n) o[i] = a[i] + b[i];
}

__global__ void k_init_h(const float* __restrict__ h0row,
                         bf16_t* __restrict__ hbuf, int n) {
  int i = blockIdx.x * blockDim.x + threadIdx.x;
  if (i < n) hbuf[i] = (bf16_t)h0row[i & (H_ - 1)];
}

__global__ void k_zero_u32(unsigned* __restrict__ p, int n) {
  int i = blockIdx.x * blockDim.x + threadIdx.x;
  if (i < n) p[i] = 0u;
}

// ---------------------------------------------------------------- GEMM
// XG[t][b][g] = sum_d A[b*T+t][d] * W[g][d] + bias[g]   (bf16 in/out, f32 acc)
// Block tile 64(M=bt) x 64(N=g), 4 waves. K stepped by 32 with DOUBLE-BUFFERED
// async global->LDS staging: next tile's copies run under current tile's WMMAs,
// drained with a partial s_wait_asynccnt.
__launch_bounds__(128)
__global__ void k_gemm_xg(const bf16_t* __restrict__ A,   // [BT, 512]
                          const bf16_t* __restrict__ W,   // [2048, 512]
                          const float*  __restrict__ bias,// [2048]
                          bf16_t* __restrict__ XG)        // [T][B][G]
{
  __shared__ bf16_t a_s[2][64][32];
  __shared__ bf16_t b_s[2][64][32];

  const int tid  = threadIdx.x;
  const int wv   = tid >> 5;
  const int lane = tid & 31;
  const int hf   = lane >> 4;
  const int l16  = lane & 15;
  const int m0   = blockIdx.x * 64;   // bt block
  const int n0   = blockIdx.y * 64;   // g  block

  const int r = tid >> 1;             // tile row this thread stages
  const int c = (tid & 1) * 16;       // element col base (0 / 16)
  const unsigned laB[2] = { lds_off(&a_s[0][0][0]) + (unsigned)tid * 32u,
                            lds_off(&a_s[1][0][0]) + (unsigned)tid * 32u };
  const unsigned lbB[2] = { lds_off(&b_s[0][0][0]) + (unsigned)tid * 32u,
                            lds_off(&b_s[1][0][0]) + (unsigned)tid * 32u };
  const bf16_t* gA = A + (size_t)(m0 + r) * 512 + c;
  const bf16_t* gW = W + (size_t)(n0 + r) * 512 + c;

  v8f acc[4];
#pragma unroll
  for (int i = 0; i < 4; ++i) acc[i] = vzero8();

  // prologue: stage K-tile 0 into buffer 0
  {
    unsigned long long ga = (unsigned long long)(const void*)gA;
    unsigned long long gb = (unsigned long long)(const void*)gW;
    ASYNC_CP16(laB[0], ga, 0);
    ASYNC_CP16(laB[0], ga, 16);
    ASYNC_CP16(lbB[0], gb, 0);
    ASYNC_CP16(lbB[0], gb, 16);
  }

  for (int p = 0; p < 16; ++p) {
    const int buf = p & 1;
    if (p + 1 < 16) {
      // stage next K-tile into the other buffer while this one computes
      unsigned long long ga =
          (unsigned long long)(const void*)(gA + (p + 1) * 32);
      unsigned long long gb =
          (unsigned long long)(const void*)(gW + (p + 1) * 32);
      ASYNC_CP16(laB[buf ^ 1], ga, 0);
      ASYNC_CP16(laB[buf ^ 1], ga, 16);
      ASYNC_CP16(lbB[buf ^ 1], gb, 0);
      ASYNC_CP16(lbB[buf ^ 1], gb, 16);
      WAIT_ASYNC(0x4);        // <=4 outstanding => tile p has landed
    } else {
      WAIT_ASYNC(0x0);
    }
    __syncthreads();          // tile p visible to all waves

    AFrag a;
#pragma unroll
    for (int e = 0; e < 16; ++e)
      a.h[e] = a_s[buf][wv * 16 + l16][amapK(e, hf)];

    AFrag b4[4];
#pragma unroll
    for (int nt = 0; nt < 4; ++nt)
#pragma unroll
      for (int e = 0; e < 16; ++e)
        b4[nt].h[e] = b_s[buf][nt * 16 + l16][bmapK(e, hf)];

#pragma unroll
    for (int nt = 0; nt < 4; ++nt)
      acc[nt] = __builtin_amdgcn_wmma_f32_16x16x32_bf16(
          false, a.v, false, b4[nt].v, (short)0, acc[nt], false, false);

    __syncthreads();          // everyone done reading before tile p+2 overwrite
  }

#pragma unroll
  for (int nt = 0; nt < 4; ++nt) {
    int g = n0 + nt * 16 + l16;
    float bi = bias[g];
    CFrag cf; cf.v = acc[nt];
#pragma unroll
    for (int rr = 0; rr < 8; ++rr) {
      int bt = m0 + wv * 16 + rr + hf * 8;
      int b  = bt >> 11;        // / T_
      int t  = bt & (T_ - 1);
      XG[((size_t)t * B_ + b) * G_ + g] = (bf16_t)(cf.f[rr] + bi);
    }
  }
}

// ---------------------------------------------------------------- persistent scan
// 16 WGs x 256 threads (8 waves). WG j owns hidden units [j*32, j*32+32).
// Whh slice (4 gates x 32 rows x 512 cols, 128KB) staged once by the Tensor
// Data Mover, then the wave's 16 B-fragments (128 VGPRs) are hoisted into
// registers for the entire 2048-step recurrence. Per step: async-stage h,
// software-pipelined A-fragment loads + WMMA, nonlinearity, h' -> global
// ping-pong, grid-wide spin barrier.
__launch_bounds__(256)
__global__ void k_scan(const bf16_t* __restrict__ XG,   // [T][B][G]
                       const bf16_t* __restrict__ Whh,  // [2048, 512]
                       const float*  __restrict__ c0row,// [H] layer slice
                       bf16_t* __restrict__ hping,      // [2][B*H] bf16
                       bf16_t* __restrict__ ybf,        // layer-0 output (or null)
                       float*  __restrict__ yf,         // layer-1 output (or null)
                       float*  __restrict__ hT,         // [B*H]
                       float*  __restrict__ cT,         // [B*H]
                       unsigned* __restrict__ syncc)
{
  extern __shared__ char smem[];
  bf16_t (*whh_s)[512] = (bf16_t(*)[512])(smem);                    // 128*512*2 = 128K
  bf16_t (*h_s)[512]   = (bf16_t(*)[512])(smem + 131072);           // 32*512*2  = 32K
  float  (*gbuf)[32]   = (float (*)[32])(smem + 131072 + 32768);    // 128*32*4  = 16K
  __shared__ float c_s[32][32];                                     // [b][hh]

  const int tid  = threadIdx.x;
  const int wv   = tid >> 5;
  const int lane = tid & 31;
  const int hf   = lane >> 4;
  const int l16  = lane & 15;
  const int j    = blockIdx.x;       // hidden-slice owner

  // ---- TDM: stage Whh slice as 3-D tile (x=512 cols, y=32 rows, z=4 gates).
  if (tid < 32) {   // one full wave issues the DMA (EXEC all ones in wave 0)
    unsigned long long ga = (unsigned long long)(const void*)Whh
                          + (unsigned long long)j * 32ull * 512ull * 2ull;
    unsigned ldsa = lds_off(&whh_s[0][0]);
    v4u g0 = { 1u,                                   // count=1, user mode
               ldsa,                                 // lds_addr
               (unsigned)(ga & 0xffffffffu),         // global_addr[31:0]
               (unsigned)((ga >> 32) & 0x01ffffffu)  // global_addr[56:32]
                 | (2u << 30) };                     // type=2 (image)
    v8i g1 = { (int)(1u << 16),        // data_size=2B; wg_mask=0
               (int)(512u << 16),      // tensor_dim0 = 512 (bits 63:48)
               (int)(512u << 16),      // tensor_dim1 = 512 (bits 95:80)
               (int)(512u << 16),      // tile_dim0 = 512 (bits 127:112)
               (int)((4u << 16) | 32u),// tile_dim1=32, tile_dim2=4
               (int)512,               // tensor_dim0_stride = 512
               0,                      // stride0 hi / stride1 lo16 (=0)
               (int)4 };               // tensor_dim1_stride = 4<<16 = 262144
    v4i g2 = { 4, 0, 262144, 0 };      // tensor_dim2=4, dim2_stride=262144
    v4i g3 = { 0, 0, 0, 0 };
#if __clang_major__ >= 23
    v8i g4 = { 0, 0, 0, 0, 0, 0, 0, 0 };
    __builtin_amdgcn_tensor_load_to_lds(g0, g1, g2, g3, g4, 0);
#else
    __builtin_amdgcn_tensor_load_to_lds(g0, g1, g2, g3, 0);
#endif
    __builtin_amdgcn_s_wait_tensorcnt((short)0);
  }
  for (int i = tid; i < 32 * 32; i += 256) {
    int b = i >> 5, hh = i & 31;
    c_s[b][hh] = c0row[j * 32 + hh];
  }
  __syncthreads();

  // ---- hoist this wave's 16 Whh B-fragments into registers (16x8 = 128 VGPRs)
  AFrag bw[16];
#pragma unroll
  for (int k = 0; k < 16; ++k)
#pragma unroll
    for (int e = 0; e < 16; ++e)
      bw[k].h[e] = whh_s[wv * 16 + l16][k * 32 + bmapK(e, hf)];

  const unsigned lhs = lds_off(&h_s[0][0]) + (unsigned)tid * 16u;

  for (int t = 0; t < T_; ++t) {
    const bf16_t* cur = hping + (size_t)(t & 1) * (B_ * H_);
    bf16_t*       nxt = hping + (size_t)((t + 1) & 1) * (B_ * H_);

    // async-stage full h [32,512] bf16 (32KB): 256 lanes x 8 x 16B
    {
      unsigned long long ga =
          (unsigned long long)(const void*)cur + (unsigned long long)tid * 16ull;
      ASYNC_CP16(lhs, ga, 0);
      ASYNC_CP16(lhs, ga, 4096);
      ASYNC_CP16(lhs, ga, 8192);
      ASYNC_CP16(lhs, ga, 12288);
      ASYNC_CP16(lhs, ga, 16384);
      ASYNC_CP16(lhs, ga, 20480);
      ASYNC_CP16(lhs, ga, 24576);
      ASYNC_CP16(lhs, ga, 28672);
      WAIT_ASYNC(0x0);
    }
    // prefetch xg gate slice 4 steps ahead toward this WGP
    if (t + 4 < T_ && tid < 128) {
      const bf16_t* pf = XG + (size_t)(t + 4) * B_ * G_
                       + (size_t)(tid >> 2) * G_ + (tid & 3) * 512 + j * 32;
      __builtin_prefetch(pf, 0, 1);
    }
    __syncthreads();

    // A-fragments software-pipelined one k ahead; B-fragments from registers
    v8f acc0 = vzero8(), acc1 = vzero8();
    AFrag a0[2], a1[2];
#pragma unroll
    for (int e = 0; e < 16; ++e) {
      int kc = amapK(e, hf);
      a0[0].h[e] = h_s[l16][kc];
      a1[0].h[e] = h_s[16 + l16][kc];
    }
#pragma unroll
    for (int k = 0; k < 16; ++k) {
      const int cb = k & 1;
      if (k + 1 < 16) {
#pragma unroll
        for (int e = 0; e < 16; ++e) {
          int kc = (k + 1) * 32 + amapK(e, hf);
          a0[cb ^ 1].h[e] = h_s[l16][kc];
          a1[cb ^ 1].h[e] = h_s[16 + l16][kc];
        }
      }
      acc0 = __builtin_amdgcn_wmma_f32_16x16x32_bf16(
          false, a0[cb].v, false, bw[k].v, (short)0, acc0, false, false);
      acc1 = __builtin_amdgcn_wmma_f32_16x16x32_bf16(
          false, a1[cb].v, false, bw[k].v, (short)0, acc1, false, false);
    }
    {
      CFrag ca; ca.v = acc0;
      CFrag cb2; cb2.v = acc1;
      int row = wv * 16 + l16;
#pragma unroll
      for (int rr = 0; rr < 8; ++rr) {
        gbuf[row][rr + hf * 8]      = ca.f[rr];
        gbuf[row][16 + rr + hf * 8] = cb2.f[rr];
      }
    }
    __syncthreads();

    // nonlinearity for our 32 hidden units x 32 batches
    const bf16_t* xgt = XG + (size_t)t * B_ * G_;
#pragma unroll
    for (int q = 0; q < 4; ++q) {
      int idx = q * 256 + tid;       // 0..1023
      int b   = idx >> 5;
      int hh  = idx & 31;
      int hidx = j * 32 + hh;
      float gi = gbuf[0 * 32 + hh][b] + (float)xgt[(size_t)b * G_ + 0 * 512 + hidx];
      float gf = gbuf[1 * 32 + hh][b] + (float)xgt[(size_t)b * G_ + 1 * 512 + hidx];
      float gg = gbuf[2 * 32 + hh][b] + (float)xgt[(size_t)b * G_ + 2 * 512 + hidx];
      float go = gbuf[3 * 32 + hh][b] + (float)xgt[(size_t)b * G_ + 3 * 512 + hidx];
      float c  = sigm(gf) * c_s[b][hh] + sigm(gi) * tanhf(gg);
      float h  = sigm(go) * tanhf(c);
      c_s[b][hh] = c;
      nxt[b * H_ + hidx] = (bf16_t)h;
      if (ybf) ybf[((size_t)b * T_ + t) * H_ + hidx] = (bf16_t)h;
      if (yf)  yf [((size_t)b * T_ + t) * H_ + hidx] = h;
      if (t == T_ - 1) {
        hT[b * H_ + hidx] = h;
        cT[b * H_ + hidx] = c;
      }
    }

    // grid-wide barrier: make h' visible, then counting barrier
    __threadfence();
    __syncthreads();
    if (tid == 0) {
      __hip_atomic_fetch_add(syncc, 1u, __ATOMIC_RELEASE, __HIP_MEMORY_SCOPE_AGENT);
      unsigned target = (unsigned)(t + 1) * (unsigned)NWG_SCAN;
      while (__hip_atomic_load(syncc, __ATOMIC_ACQUIRE, __HIP_MEMORY_SCOPE_AGENT) < target) {
        __builtin_amdgcn_s_sleep(2);
      }
    }
    __syncthreads();
  }
}

// ---------------------------------------------------------------- launch
extern "C" void kernel_launch(void* const* d_in, const int* in_sizes, int n_in,
                              void* d_out, int out_size, void* d_ws, size_t ws_size,
                              hipStream_t stream) {
  (void)in_sizes; (void)n_in; (void)out_size; (void)ws_size;

  const float* x    = (const float*)d_in[0];
  const float* Wih0 = (const float*)d_in[1];
  const float* Whh0 = (const float*)d_in[2];
  const float* bih0 = (const float*)d_in[3];
  const float* bhh0 = (const float*)d_in[4];
  const float* Wih1 = (const float*)d_in[5];
  const float* Whh1 = (const float*)d_in[6];
  const float* bih1 = (const float*)d_in[7];
  const float* bhh1 = (const float*)d_in[8];
  const float* h0   = (const float*)d_in[9];
  const float* c0   = (const float*)d_in[10];

  char* ws = (char*)d_ws;
  size_t off = 0;
  bf16_t* xg  = (bf16_t*)(ws + off); off += (size_t)BT_ * G_ * 2;   // 256 MB
  bf16_t* xbf = (bf16_t*)(ws + off); off += (size_t)BT_ * D_ * 2;   // 64 MB (x, then y0)
  bf16_t* w0i = (bf16_t*)(ws + off); off += (size_t)G_ * D_ * 2;
  bf16_t* w0h = (bf16_t*)(ws + off); off += (size_t)G_ * H_ * 2;
  bf16_t* w1i = (bf16_t*)(ws + off); off += (size_t)G_ * H_ * 2;
  bf16_t* w1h = (bf16_t*)(ws + off); off += (size_t)G_ * H_ * 2;
  float* bias0 = (float*)(ws + off); off += (size_t)G_ * 4;
  float* bias1 = (float*)(ws + off); off += (size_t)G_ * 4;
  bf16_t* hping = (bf16_t*)(ws + off); off += (size_t)2 * B_ * H_ * 2;
  unsigned* syncc = (unsigned*)(ws + off); off += 256;

  float* out_y = (float*)d_out;
  float* out_h = out_y + (size_t)B_ * T_ * H_;
  float* out_c = out_h + (size_t)L_ * B_ * H_;

  // conversions to bf16 + fused bias
  {
    int n = BT_ * D_;
    k_cvt_bf16<<<(n + 255) / 256, 256, 0, stream>>>(x, xbf, n);
    n = G_ * D_;
    k_cvt_bf16<<<(n + 255) / 256, 256, 0, stream>>>(Wih0, w0i, n);
    n = G_ * H_;
    k_cvt_bf16<<<(n + 255) / 256, 256, 0, stream>>>(Whh0, w0h, n);
    k_cvt_bf16<<<(n + 255) / 256, 256, 0, stream>>>(Wih1, w1i, n);
    k_cvt_bf16<<<(n + 255) / 256, 256, 0, stream>>>(Whh1, w1h, n);
    k_bias_sum<<<(G_ + 255) / 256, 256, 0, stream>>>(bih0, bhh0, bias0, G_);
    k_bias_sum<<<(G_ + 255) / 256, 256, 0, stream>>>(bih1, bhh1, bias1, G_);
  }

  dim3 ggrid(BT_ / 64, G_ / 64);
  size_t smem = 131072 + 32768 + 16384;   // Whh slice + h stage + gate buf

  // ---- layer 0
  k_gemm_xg<<<ggrid, 128, 0, stream>>>(xbf, w0i, bias0, xg);
  k_zero_u32<<<1, 32, 0, stream>>>(syncc, 8);
  k_init_h<<<(B_ * H_ + 255) / 256, 256, 0, stream>>>(h0, hping, B_ * H_);
  k_scan<<<NWG_SCAN, 256, smem, stream>>>(xg, w0h, c0, hping,
                                          /*ybf=*/xbf, /*yf=*/nullptr,
                                          out_h, out_c, syncc);

  // ---- layer 1
  k_gemm_xg<<<ggrid, 128, 0, stream>>>(xbf, w1i, bias1, xg);
  k_zero_u32<<<1, 32, 0, stream>>>(syncc, 8);
  k_init_h<<<(B_ * H_ + 255) / 256, 256, 0, stream>>>(h0 + H_, hping, B_ * H_);
  k_scan<<<NWG_SCAN, 256, smem, stream>>>(xg, w1h, c0 + H_, hping,
                                          /*ybf=*/nullptr, /*yf=*/out_y,
                                          out_h + (size_t)B_ * H_,
                                          out_c + (size_t)B_ * H_, syncc);
}
